// Block_88252987998566
// MI455X (gfx1250) — matrix operand order repaired and checked by reading, not compiled, not run on hardware
//
#include <hip/hip_runtime.h>
#include <hip/hip_bf16.h>
#include <math.h>

// ---------------------------------------------------------------------------
// Transformer block forward for MI455X (gfx1250, wave32, WMMA + TDM).
// Shapes fixed by the reference: B=4, T=2048, D=1024, H=16, HD=64.
// GEMMs: v_wmma_f32_16x16x32_bf16 with LDS-staged tiles via tensor_load_to_lds
// (TDM DMA, double-buffered, TENSORcnt-tracked).
// ---------------------------------------------------------------------------

typedef __bf16 v16bf __attribute__((ext_vector_type(16)));
typedef __bf16 v8bf  __attribute__((ext_vector_type(8)));
typedef float  v8f   __attribute__((ext_vector_type(8)));
typedef int    si4   __attribute__((ext_vector_type(4)));
typedef int    si8   __attribute__((ext_vector_type(8)));

#define BB  4
#define TT  2048
#define DD  1024
#define HH  16
#define HDD 64
#define DFF 4096
#define ROWS (BB*TT)          // 8192
#define LN_EPS 1e-5f

static __device__ __forceinline__ v8f v8f_zero() {
  v8f z;
#pragma unroll
  for (int i = 0; i < 8; ++i) z[i] = 0.0f;
  return z;
}

// A-fragment (16x32 bf16): lane L holds row M = L%16; element e maps to
// K = 16*(L>>4) + e.  p must point at row base + k0 + 16*(L>>4).
static __device__ __forceinline__ v16bf load_a_frag(const __bf16* p) {
  union { v16bf v; v8bf h[2]; } u;
  u.h[0] = *(const v8bf*)(p);
  u.h[1] = *(const v8bf*)(p + 8);
  return u.v;
}

static __device__ __forceinline__ v8f wmma_bf16(v16bf a, v16bf b, v8f c) {
  return __builtin_amdgcn_wmma_f32_16x16x32_bf16(false, a, false, b,
                                                 (short)0, c, false, false);
}

// Flat address of a __shared__ object: low 32 bits are the LDS byte offset.
static __device__ __forceinline__ unsigned lds_offset_of(const void* p) {
  return (unsigned)(unsigned long long)(uintptr_t)p;
}

// ---------------------------------------------------------------------------
// Tensor Data Mover: DMA a 2-D bf16 tile (rows x kelems, row stride ldk
// elements) from global memory into LDS at lds_off.  Issued once per wave
// (EXEC-independent); tracked on TENSORcnt.  D# layout per CDNA5 ISA 8.3/8.4.
// ---------------------------------------------------------------------------
static __device__ __forceinline__ void tdm_load_tile(
    const __bf16* gsrc, unsigned lds_off, int kelems, int rows, int ldk) {
  const unsigned long long ga = (unsigned long long)(uintptr_t)gsrc;
  si4 g0;
  g0[0] = 1;                                   // count=1, user descriptor
  g0[1] = (int)lds_off;                        // lds_addr (bytes)
  g0[2] = (int)(unsigned)(ga & 0xFFFFFFFFull); // global_addr[31:0]
  g0[3] = (int)((unsigned)((ga >> 32) & 0x1FFFFFFull) | 0x80000000u); // type=2
  si8 g1;
  g1[0] = 0x00010000;                          // wg_mask=0, data_size=1 (2B)
  g1[1] = (int)((unsigned)(kelems & 0xFFFF) << 16);        // tensor_dim0 lo
  g1[2] = (int)(((unsigned)kelems >> 16) & 0xFFFF) |
          (int)((unsigned)(rows & 0xFFFF) << 16);          // td0 hi | td1 lo
  g1[3] = (int)(((unsigned)rows >> 16) & 0xFFFF) |
          (int)((unsigned)(kelems & 0xFFFF) << 16);        // td1 hi | tile_dim0
  g1[4] = rows & 0xFFFF;                       // tile_dim1 | tile_dim2=0
  g1[5] = ldk;                                 // tensor_dim0_stride[31:0]
  g1[6] = 0;                                   // stride hi | td1_stride lo
  g1[7] = 0;
  asm volatile("tensor_load_to_lds %0, %1" :: "s"(g0), "s"(g1) : "memory");
}

// ---------------------------------------------------------------------------
// Weight transpose + f32 -> bf16 cast:  W[K][N] -> Wt[N][K]
// ---------------------------------------------------------------------------
__global__ __launch_bounds__(256) void wtrans_kernel(
    const float* __restrict__ W, __bf16* __restrict__ Wt, int K, int N) {
  size_t idx = (size_t)blockIdx.x * 256 + threadIdx.x;
  if (idx >= (size_t)K * N) return;
  int k = (int)(idx / N);
  int n = (int)(idx % N);
  Wt[(size_t)n * K + k] = (__bf16)W[idx];
}

// ---------------------------------------------------------------------------
// LayerNorm (one 256-thread block per row of 1024), f32 in -> bf16 out
// ---------------------------------------------------------------------------
__global__ __launch_bounds__(256) void ln_kernel(
    const float* __restrict__ x, const float* __restrict__ g,
    const float* __restrict__ be, __bf16* __restrict__ out) {
  __shared__ float red[256];
  const int tid = threadIdx.x;
  const size_t row = blockIdx.x;
  const float* xr = x + row * DD;

  float v[4];
  float s = 0.f;
#pragma unroll
  for (int j = 0; j < 4; ++j) { v[j] = xr[tid + 256 * j]; s += v[j]; }
  red[tid] = s; __syncthreads();
  for (int off = 128; off > 0; off >>= 1) {
    if (tid < off) red[tid] += red[tid + off];
    __syncthreads();
  }
  const float mu = red[0] * (1.0f / DD);
  __syncthreads();

  float s2 = 0.f;
#pragma unroll
  for (int j = 0; j < 4; ++j) { float d = v[j] - mu; s2 += d * d; }
  red[tid] = s2; __syncthreads();
  for (int off = 128; off > 0; off >>= 1) {
    if (tid < off) red[tid] += red[tid + off];
    __syncthreads();
  }
  const float rstd = rsqrtf(red[0] * (1.0f / DD) + LN_EPS);

#pragma unroll
  for (int j = 0; j < 4; ++j) {
    int c = tid + 256 * j;
    out[row * DD + c] = (__bf16)((v[j] - mu) * rstd * g[c] + be[c]);
  }
}

// ---------------------------------------------------------------------------
// WMMA GEMM: C[M,N] = A[M,K] * Bt[N,K]^T   (A, Bt bf16 row-major)
// 256 threads = 8 waves; wave tile 32x64; block tile 128x128.
// A and B 128x32 k-slices are DMA'd into double-buffered LDS by wave 0 via
// tensor_load_to_lds; all waves read fragments with ds_load_b128.
// mode 1: bf16 out scattered to [B,H,T,HD]          (Q, K projections)
// mode 2: bf16 out scattered to [B,H,HD,T]          (V projection, transposed)
// mode 3: f32 out = acc + bias[n] + res[m*N+n]      (O projection + residual)
// mode 4: bf16 out = relu(acc + bias[n])            (FF1)
// mode 5: f32 out = res[m*N+n] + relu(acc+bias[n])  (FF2 -> final output)
// ---------------------------------------------------------------------------
__global__ __launch_bounds__(256) void gemm_wmma_kernel(
    const __bf16* __restrict__ A, const __bf16* __restrict__ Bt,
    int M, int N, int K, int mode,
    const float* __restrict__ bias, const float* __restrict__ res,
    float* __restrict__ outf, __bf16* __restrict__ outb) {
  __shared__ __align__(32) __bf16 ldsA[2][128 * 32];   // 2 x 8 KB
  __shared__ __align__(32) __bf16 ldsB[2][128 * 32];   // 2 x 8 KB

  const int lane = threadIdx.x & 31;
  const int wid  = threadIdx.x >> 5;
  const int hsel = lane >> 4;
  const int l15  = lane & 15;
  const int mw   = wid & 3;
  const int nw   = wid >> 2;
  const int mblk = blockIdx.y * 128;
  const int nblk = blockIdx.x * 128;
  const int m0   = mblk + mw * 32;
  const int n0   = nblk + nw * 64;

  v8f acc[2][4];
#pragma unroll
  for (int i = 0; i < 2; ++i)
#pragma unroll
    for (int j = 0; j < 4; ++j) acc[i][j] = v8f_zero();

  // Prologue: DMA first k-slice (buffer 0).
  if (wid == 0) {
    tdm_load_tile(A  + (size_t)mblk * K, lds_offset_of(&ldsA[0][0]), 32, 128, K);
    tdm_load_tile(Bt + (size_t)nblk * K, lds_offset_of(&ldsB[0][0]), 32, 128, K);
  }

  for (int k0 = 0; k0 < K; k0 += 32) {
    const int buf = (k0 >> 5) & 1;
    if (wid == 0) {
      if (k0 + 32 < K) {
        // Kick off next slice into the other buffer, then wait until the
        // current pair (oldest two TDM ops, in-order completion) is done.
        tdm_load_tile(A  + (size_t)mblk * K + k0 + 32,
                      lds_offset_of(&ldsA[buf ^ 1][0]), 32, 128, K);
        tdm_load_tile(Bt + (size_t)nblk * K + k0 + 32,
                      lds_offset_of(&ldsB[buf ^ 1][0]), 32, 128, K);
        __builtin_amdgcn_s_wait_tensorcnt(2);
      } else {
        __builtin_amdgcn_s_wait_tensorcnt(0);
      }
    }
    __syncthreads();   // current buffer ready for all waves

    const __bf16* la = &ldsA[buf][0];
    const __bf16* lb = &ldsB[buf][0];
    v16bf a0 = load_a_frag(la + (size_t)(mw * 32 + l15)      * 32 + 16 * hsel);
    v16bf a1 = load_a_frag(la + (size_t)(mw * 32 + 16 + l15) * 32 + 16 * hsel);
#pragma unroll
    for (int j = 0; j < 4; ++j) {
      v16bf b = *(const v16bf*)(lb + (size_t)(nw * 64 + 16 * j + l15) * 32 +
                                16 * hsel);
      acc[0][j] = wmma_bf16(a0, b, acc[0][j]);
      acc[1][j] = wmma_bf16(a1, b, acc[1][j]);
    }
    __syncthreads();   // all reads done before this buffer is overwritten
  }

  // Epilogue. C layout: VGPR r, lane L -> (M = r + 8*(L>>4), N = L&15).
#pragma unroll
  for (int i = 0; i < 2; ++i) {
#pragma unroll
    for (int j = 0; j < 4; ++j) {
#pragma unroll
      for (int r = 0; r < 8; ++r) {
        const int m = m0 + 16 * i + r + 8 * hsel;
        const int n = n0 + 16 * j + l15;
        const float v = acc[i][j][r];
        if (mode == 1) {                       // [B,H,T,HD]
          int b = m >> 11, t = m & (TT - 1);
          int hh = n >> 6, hd = n & (HDD - 1);
          outb[(((size_t)(b * HH + hh) * TT) + t) * HDD + hd] = (__bf16)v;
        } else if (mode == 2) {                // [B,H,HD,T]
          int b = m >> 11, t = m & (TT - 1);
          int hh = n >> 6, hd = n & (HDD - 1);
          outb[(((size_t)(b * HH + hh) * HDD) + hd) * TT + t] = (__bf16)v;
        } else if (mode == 3) {
          outf[(size_t)m * N + n] = v + bias[n] + res[(size_t)m * N + n];
        } else if (mode == 4) {
          float o = v + bias[n];
          outb[(size_t)m * N + n] = (__bf16)(o > 0.f ? o : 0.f);
        } else {                               // mode 5
          float o = v + bias[n];
          o = o > 0.f ? o : 0.f;
          outf[(size_t)m * N + n] = res[(size_t)m * N + n] + o;
        }
      }
    }
  }
}

// ---------------------------------------------------------------------------
// Causal flash attention.
// Q,K: bf16 [B*H, T, 64]; Vt: bf16 [B*H, 64, T]; O: bf16 [B, T, D].
// 128 threads = 4 waves; each wave owns 16 q-rows; block covers 64 q-rows.
// ---------------------------------------------------------------------------
__global__ __launch_bounds__(128) void attn_kernel(
    const __bf16* __restrict__ Q, const __bf16* __restrict__ Kc,
    const __bf16* __restrict__ Vt, __bf16* __restrict__ O) {
  __shared__ __align__(16) __bf16 pl[4][16][80];   // padded: 160B row stride

  const int lane = threadIdx.x & 31;
  const int w    = threadIdx.x >> 5;
  const int hsel = lane >> 4;
  const int l15  = lane & 15;
  const int bh   = blockIdx.x;          // 0..63  (b*16 + h)
  const int by   = blockIdx.y;          // 0..31  (64-row q tile)
  const int q0   = by * 64 + w * 16;

  const __bf16* Qb = Q  + (size_t)bh * TT * HDD;
  const __bf16* Kb = Kc + (size_t)bh * TT * HDD;
  const __bf16* Vb = Vt + (size_t)bh * HDD * TT;

  v16bf aq[2];
  aq[0] = load_a_frag(Qb + (size_t)(q0 + l15) * HDD + 0  + 16 * hsel);
  aq[1] = load_a_frag(Qb + (size_t)(q0 + l15) * HDD + 32 + 16 * hsel);

  v8f o[4];
#pragma unroll
  for (int nt = 0; nt < 4; ++nt) o[nt] = v8f_zero();
  float mrun[8], lrun[8];
#pragma unroll
  for (int r = 0; r < 8; ++r) { mrun[r] = -__builtin_inff(); lrun[r] = 0.f; }

  for (int kt = 0; kt <= by; ++kt) {
    // ---- S = Q K^T for 64 keys (4 N-tiles, K-dim = 64 -> 2 wmma steps) ----
    v8f s[4];
#pragma unroll
    for (int nt = 0; nt < 4; ++nt) {
      s[nt] = v8f_zero();
      const int kn = kt * 64 + nt * 16;
#pragma unroll
      for (int ks = 0; ks < 2; ++ks) {
        v16bf b = *(const v16bf*)(Kb + (size_t)(kn + l15) * HDD +
                                  32 * ks + 16 * hsel);
        s[nt] = wmma_bf16(aq[ks], b, s[nt]);
      }
    }
    // ---- scale + causal mask (only last tile can cross the diagonal) ----
    const bool masked = (kt == by);
#pragma unroll
    for (int nt = 0; nt < 4; ++nt) {
#pragma unroll
      for (int r = 0; r < 8; ++r) {
        float v = s[nt][r] * 0.125f;                 // 1/sqrt(64)
        if (masked) {
          int key = kt * 64 + nt * 16 + l15;
          int qq  = q0 + r + 8 * hsel;
          if (key > qq) v = -__builtin_inff();
        }
        s[nt][r] = v;
      }
    }
    // ---- online softmax: per-row stats via 16-lane shfl reductions ----
    float mnew[8];
#pragma unroll
    for (int r = 0; r < 8; ++r) {
      float t = s[0][r];
#pragma unroll
      for (int nt = 1; nt < 4; ++nt) t = fmaxf(t, s[nt][r]);
      t = fmaxf(t, __shfl_xor(t, 1, 32));
      t = fmaxf(t, __shfl_xor(t, 2, 32));
      t = fmaxf(t, __shfl_xor(t, 4, 32));
      t = fmaxf(t, __shfl_xor(t, 8, 32));
      mnew[r] = fmaxf(mrun[r], t);
    }
#pragma unroll
    for (int r = 0; r < 8; ++r) {
      float rs = 0.f;
#pragma unroll
      for (int nt = 0; nt < 4; ++nt) {
        float p = __expf(s[nt][r] - mnew[r]);
        rs += p;
        pl[w][r + 8 * hsel][nt * 16 + l15] = (__bf16)p;
      }
      rs += __shfl_xor(rs, 1, 32);
      rs += __shfl_xor(rs, 2, 32);
      rs += __shfl_xor(rs, 4, 32);
      rs += __shfl_xor(rs, 8, 32);
      const float alpha = __expf(mrun[r] - mnew[r]);
      lrun[r] = lrun[r] * alpha + rs;
      mrun[r] = mnew[r];
#pragma unroll
      for (int nt = 0; nt < 4; ++nt) o[nt][r] *= alpha;
    }
    // ---- reload P in A-fragment layout from LDS (same-wave, DS in-order) --
    v16bf ap0 = load_a_frag(&pl[w][l15][0  + 16 * hsel]);
    v16bf ap1 = load_a_frag(&pl[w][l15][32 + 16 * hsel]);
    // ---- O += P * V  (V pre-transposed: lane = hd row, contiguous keys) --
#pragma unroll
    for (int nt = 0; nt < 4; ++nt) {
      const __bf16* vrow = Vb + (size_t)(nt * 16 + l15) * TT + kt * 64;
      v16bf b0 = *(const v16bf*)(vrow + 0  + 16 * hsel);
      v16bf b1 = *(const v16bf*)(vrow + 32 + 16 * hsel);
      o[nt] = wmma_bf16(ap0, b0, o[nt]);
      o[nt] = wmma_bf16(ap1, b1, o[nt]);
    }
  }

  // ---- normalize and scatter to [B, T, D] as bf16 ----
  const int b  = bh >> 4;
  const int hh = bh & 15;
#pragma unroll
  for (int r = 0; r < 8; ++r) {
    const float inv = 1.0f / lrun[r];
    const int q = q0 + r + 8 * hsel;
#pragma unroll
    for (int nt = 0; nt < 4; ++nt) {
      const int d = hh * HDD + nt * 16 + l15;
      O[((size_t)(b * TT + q)) * DD + d] = (__bf16)(o[nt][r] * inv);
    }
  }
}

// ---------------------------------------------------------------------------
// Host-side orchestration
// ---------------------------------------------------------------------------
extern "C" void kernel_launch(void* const* d_in, const int* in_sizes, int n_in,
                              void* d_out, int out_size, void* d_ws,
                              size_t ws_size, hipStream_t stream) {
  const float* x    = (const float*)d_in[0];
  const float* Wq   = (const float*)d_in[1];
  const float* Wk   = (const float*)d_in[2];
  const float* Wv   = (const float*)d_in[3];
  const float* Wo   = (const float*)d_in[4];
  const float* bo   = (const float*)d_in[5];
  const float* W1   = (const float*)d_in[6];
  const float* b1   = (const float*)d_in[7];
  const float* W2   = (const float*)d_in[8];
  const float* b2   = (const float*)d_in[9];
  const float* ln1g = (const float*)d_in[10];
  const float* ln1b = (const float*)d_in[11];
  const float* ln2g = (const float*)d_in[12];
  const float* ln2b = (const float*)d_in[13];

  char* ws = (char*)d_ws;
  const size_t MB = 1ull << 20;
  __bf16* Wqt  = (__bf16*)(ws + 0 * MB);     // 2 MB
  __bf16* Wkt  = (__bf16*)(ws + 2 * MB);     // 2 MB
  __bf16* Wvt  = (__bf16*)(ws + 4 * MB);     // 2 MB
  __bf16* Wot  = (__bf16*)(ws + 6 * MB);     // 2 MB
  __bf16* W1t  = (__bf16*)(ws + 8 * MB);     // 8 MB  [4096,1024]
  __bf16* W2t  = (__bf16*)(ws + 16 * MB);    // 8 MB  [1024,4096]
  __bf16* h1   = (__bf16*)(ws + 24 * MB);    // 16 MB [8192,1024]
  __bf16* qb   = (__bf16*)(ws + 40 * MB);    // 16 MB [B,H,T,HD]
  __bf16* kb   = (__bf16*)(ws + 56 * MB);    // 16 MB [B,H,T,HD]
  __bf16* vtb  = (__bf16*)(ws + 72 * MB);    // 16 MB [B,H,HD,T]
  __bf16* atnb = (__bf16*)(ws + 88 * MB);    // 16 MB [B,T,D]
  float*  x2   = (float*)(ws + 104 * MB);    // 32 MB [8192,1024]
  __bf16* h2   = (__bf16*)(ws + 136 * MB);   // 16 MB [8192,1024]
  __bf16* ffb  = (__bf16*)(ws + 152 * MB);   // 64 MB [8192,4096]

  // 1) Weights: transpose + cast to bf16 (Wt[N][K])
  {
    int n1 = DD * DD;
    wtrans_kernel<<<(n1 + 255) / 256, 256, 0, stream>>>(Wq, Wqt, DD, DD);
    wtrans_kernel<<<(n1 + 255) / 256, 256, 0, stream>>>(Wk, Wkt, DD, DD);
    wtrans_kernel<<<(n1 + 255) / 256, 256, 0, stream>>>(Wv, Wvt, DD, DD);
    wtrans_kernel<<<(n1 + 255) / 256, 256, 0, stream>>>(Wo, Wot, DD, DD);
    int n2 = DD * DFF;
    wtrans_kernel<<<(n2 + 255) / 256, 256, 0, stream>>>(W1, W1t, DD, DFF);
    wtrans_kernel<<<(n2 + 255) / 256, 256, 0, stream>>>(W2, W2t, DFF, DD);
  }

  // 2) h1 = LN1(x)  (bf16)
  ln_kernel<<<ROWS, 256, 0, stream>>>(x, ln1g, ln1b, h1);

  // 3) Q/K/V projections
  dim3 gqkv(DD / 128, ROWS / 128);   // (8, 64)
  gemm_wmma_kernel<<<gqkv, 256, 0, stream>>>(h1, Wqt, ROWS, DD, DD, 1,
                                             nullptr, nullptr, nullptr, qb);
  gemm_wmma_kernel<<<gqkv, 256, 0, stream>>>(h1, Wkt, ROWS, DD, DD, 1,
                                             nullptr, nullptr, nullptr, kb);
  gemm_wmma_kernel<<<gqkv, 256, 0, stream>>>(h1, Wvt, ROWS, DD, DD, 2,
                                             nullptr, nullptr, nullptr, vtb);

  // 4) Causal attention -> atnb [B,T,D] bf16
  attn_kernel<<<dim3(BB * HH, TT / 64), 128, 0, stream>>>(qb, kb, vtb, atnb);

  // 5) O projection + bias + residual -> x2 (f32)
  gemm_wmma_kernel<<<gqkv, 256, 0, stream>>>(atnb, Wot, ROWS, DD, DD, 3,
                                             bo, x, x2, nullptr);

  // 6) h2 = LN2(x2)  (bf16)
  ln_kernel<<<ROWS, 256, 0, stream>>>(x2, ln2g, ln2b, h2);

  // 7) FF1: relu(h2 W1 + b1) -> ffb (bf16)
  dim3 gff1(DFF / 128, ROWS / 128);  // (32, 64)
  gemm_wmma_kernel<<<gff1, 256, 0, stream>>>(h2, W1t, ROWS, DFF, DD, 4,
                                             b1, nullptr, nullptr, ffb);

  // 8) FF2: x2 + relu(ffb W2 + b2) -> d_out (f32)
  dim3 gff2(DD / 128, ROWS / 128);   // (8, 64)
  gemm_wmma_kernel<<<gff2, 256, 0, stream>>>(ffb, W2t, ROWS, DD, DFF, 5,
                                             b2, x2, (float*)d_out, nullptr);
}
